// GAT_63608465653856
// MI455X (gfx1250) — matrix operand order repaired and checked by reading, not compiled, not run on hardware
//
#include <hip/hip_runtime.h>
#include <hip/hip_bf16.h>
#include <math.h>

typedef __attribute__((ext_vector_type(16))) _Float16 v16h;
typedef __attribute__((ext_vector_type(8)))  float    v8f;

#define NN      50000
#define NE      600000
#define DIM     128
#define HEADS   8
#define CHEAD   16
#define OUTD    64
#define NEG_SLOPE 0.2f

// ---------------------------------------------------------------------------
// Weight pre-swizzle: f32 [128 x ncols] row-major -> f16 fragments laid out as
// [kb (K/32)][ct (ncols/16)][lane 0..31][i 0..15] so a wave loads one B frag
// (v_wmma_f32_16x16x32_f16 B operand, 32x16) as a single contiguous v16h/lane.
// Element i at lane:  K = 32*kb + 16*(lane>>4) + i,  N = 16*ct + (lane&15).
// ---------------------------------------------------------------------------
__global__ void swizzle_w(const float* __restrict__ W,
                          unsigned short* __restrict__ outw, int ncols) {
  int nct = ncols >> 4;
  int total = 4 * nct * 32 * 16;
  int t = blockIdx.x * blockDim.x + threadIdx.x;
  if (t >= total) return;
  int i    = t & 15;
  int lane = (t >> 4) & 31;
  int ct   = (t >> 9) % nct;
  int kb   = t / (512 * nct);
  int k = 32 * kb + 16 * (lane >> 4) + i;
  int n = 16 * ct + (lane & 15);
  _Float16 hv = (_Float16)W[k * ncols + n];
  outw[t] = *(const unsigned short*)&hv;
}

// ---------------------------------------------------------------------------
// Fused node GEMM: x[N,128] x {W_lin, W_src, W_dst} -> lin(+bias), xs, xd.
// Block = 256 threads = 8 waves; block owns one 16-row M tile; wave w owns
// column tile ct=w of each of the 3 matrices (A fragment reused 3x per K step).
// ---------------------------------------------------------------------------
__global__ void __launch_bounds__(256) node_gemm(
    const float* __restrict__ x, const unsigned short* __restrict__ wswz,
    const float* __restrict__ bias_lin,
    float* __restrict__ lin, float* __restrict__ xs, float* __restrict__ xd) {
  const int lane = threadIdx.x & 31;
  const int wave = threadIdx.x >> 5;          // 0..7 == column tile
  const int lo = lane & 15, hi = lane >> 4;
  const int row_m = blockIdx.x * 16;

  v8f acc0 = {}, acc1 = {}, acc2 = {};
  const float* arow = x + (size_t)(row_m + lo) * DIM;
  const v16h* bfr = (const v16h*)wswz;        // 1024 v16h per matrix

#pragma unroll
  for (int kb = 0; kb < 4; ++kb) {
    // A fragment (16x32 f16): lane holds row lo; halves 0..15 = K 32kb+16hi..,
    // (contiguous, matching the swizzled B layout)
    v16h a;
    const float* p0 = arow + 32 * kb + 16 * hi;
#pragma unroll
    for (int j = 0; j < 8; ++j) a[j] = (_Float16)p0[j];
#pragma unroll
    for (int j = 0; j < 8; ++j) a[8 + j] = (_Float16)p0[8 + j];

    v16h b0 = bfr[0 * 1024 + (kb * 8 + wave) * 32 + lane];
    v16h b1 = bfr[1 * 1024 + (kb * 8 + wave) * 32 + lane];
    v16h b2 = bfr[2 * 1024 + (kb * 8 + wave) * 32 + lane];

    acc0 = __builtin_amdgcn_wmma_f32_16x16x32_f16(false, a, false, b0, (short)0, acc0, false, false);
    acc1 = __builtin_amdgcn_wmma_f32_16x16x32_f16(false, a, false, b1, (short)0, acc1, false, false);
    acc2 = __builtin_amdgcn_wmma_f32_16x16x32_f16(false, a, false, b2, (short)0, acc2, false, false);
  }

  const int col = wave * 16 + lo;             // fixed per lane across C frag
  const float bl = bias_lin[col];
#pragma unroll
  for (int r = 0; r < 8; ++r) {
    int row = row_m + hi * 8 + r;             // C/D layout: VGPR r, half 'hi'
    size_t o = (size_t)row * DIM + col;
    lin[o] = acc0[r] + bl;
    xs[o]  = acc1[r];
    xd[o]  = acc2[r];
  }
}

// Output GEMM: h[N,128] @ W_out[128,64] + b_out. 4 waves per block, ct=wave.
__global__ void __launch_bounds__(128) out_gemm(
    const float* __restrict__ h, const unsigned short* __restrict__ wswz,
    const float* __restrict__ bias, float* __restrict__ out) {
  const int lane = threadIdx.x & 31;
  const int wave = threadIdx.x >> 5;          // 0..3
  const int lo = lane & 15, hi = lane >> 4;
  const int row_m = blockIdx.x * 16;

  v8f acc = {};
  const float* arow = h + (size_t)(row_m + lo) * DIM;
  const v16h* bfr = (const v16h*)wswz;
#pragma unroll
  for (int kb = 0; kb < 4; ++kb) {
    v16h a;
    const float* p0 = arow + 32 * kb + 16 * hi;
#pragma unroll
    for (int j = 0; j < 8; ++j) a[j] = (_Float16)p0[j];
#pragma unroll
    for (int j = 0; j < 8; ++j) a[8 + j] = (_Float16)p0[8 + j];
    v16h b = bfr[(kb * 4 + wave) * 32 + lane];
    acc = __builtin_amdgcn_wmma_f32_16x16x32_f16(false, a, false, b, (short)0, acc, false, false);
  }
  const int col = wave * 16 + lo;
  const float bo = bias[col];
#pragma unroll
  for (int r = 0; r < 8; ++r) {
    int row = row_m + hi * 8 + r;
    out[(size_t)row * OUTD + col] = acc[r] + bo;
  }
}

// Per-(node,head) attention logits: alpha = sum_c xs[n,h,c]*att[h,c]
__global__ void alpha_kernel(const float* __restrict__ xs, const float* __restrict__ xd,
                             const float* __restrict__ att_s, const float* __restrict__ att_d,
                             float* __restrict__ as_, float* __restrict__ ad_) {
  int t = blockIdx.x * blockDim.x + threadIdx.x;
  if (t >= NN * HEADS) return;
  int h = t & 7, n = t >> 3;
  const float* ps = xs + (size_t)n * DIM + h * CHEAD;
  const float* pd = xd + (size_t)n * DIM + h * CHEAD;
  const float* vs = att_s + h * CHEAD;
  const float* vd = att_d + h * CHEAD;
  float ss = 0.f, sd = 0.f;
#pragma unroll
  for (int c = 0; c < CHEAD; ++c) { ss += ps[c] * vs[c]; sd += pd[c] * vd[c]; }
  as_[t] = ss;
  ad_[t] = sd;
}

__global__ void init_buffers(float* __restrict__ m, float* __restrict__ denom,
                             float* __restrict__ agg) {
  int t = blockIdx.x * blockDim.x + threadIdx.x;
  if (t >= NN * DIM) return;
  agg[t] = 0.f;
  if (t < NN * HEADS) {
    m[t] = -__builtin_inff();
    denom[t] = 0.f;
  }
}

__device__ __forceinline__ float atomic_max_f32(float* addr, float val) {
  if (val >= 0.0f)
    return __int_as_float(atomicMax((int*)addr, __float_as_int(val)));
  return __uint_as_float(atomicMin((unsigned int*)addr, __float_as_uint(val)));
}

__device__ __forceinline__ float leaky(float v) {
  return v > 0.f ? v : NEG_SLOPE * v;
}

// Segment max over destination nodes
__global__ void edge_max(const int* __restrict__ ei, const float* __restrict__ as_,
                         const float* __restrict__ ad_, float* __restrict__ m) {
  int e = blockIdx.x * blockDim.x + threadIdx.x;
  if (e >= NE) return;
  int s = ei[e], d = ei[NE + e];
#pragma unroll
  for (int h = 0; h < HEADS; ++h) {
    float v = leaky(as_[s * HEADS + h] + ad_[d * HEADS + h]);
    atomic_max_f32(&m[d * HEADS + h], v);
  }
}

// p = exp(e - m[dst]); segment-sum denominators; cache p per (edge,head)
__global__ void edge_sum(const int* __restrict__ ei, const float* __restrict__ as_,
                         const float* __restrict__ ad_, const float* __restrict__ m,
                         float* __restrict__ denom, float* __restrict__ pbuf) {
  int e = blockIdx.x * blockDim.x + threadIdx.x;
  if (e >= NE) return;
  int s = ei[e], d = ei[NE + e];
#pragma unroll
  for (int h = 0; h < HEADS; ++h) {
    float v = leaky(as_[s * HEADS + h] + ad_[d * HEADS + h]);
    float p = __expf(v - m[d * HEADS + h]);
    pbuf[e * HEADS + h] = p;
    atomicAdd(&denom[d * HEADS + h], p);
  }
}

// agg[dst] += (p/denom) * xs[src]   (one thread per (edge, head))
__global__ void edge_agg(const int* __restrict__ ei, const float* __restrict__ pbuf,
                         const float* __restrict__ denom, const float* __restrict__ xs,
                         float* __restrict__ agg) {
  int t = blockIdx.x * blockDim.x + threadIdx.x;
  if (t >= NE * HEADS) return;
  int h = t & 7, e = t >> 3;
  int s = ei[e], d = ei[NE + e];
  float alpha = pbuf[t] / (denom[d * HEADS + h] + 1e-16f);
  const float4* src = (const float4*)(xs + (size_t)s * DIM + h * CHEAD);
  float* dst = agg + (size_t)d * DIM + h * CHEAD;
#pragma unroll
  for (int q = 0; q < 4; ++q) {
    float4 v = src[q];
    atomicAdd(dst + 4 * q + 0, alpha * v.x);
    atomicAdd(dst + 4 * q + 1, alpha * v.y);
    atomicAdd(dst + 4 * q + 2, alpha * v.z);
    atomicAdd(dst + 4 * q + 3, alpha * v.w);
  }
}

// h = elu(lin + agg + b_conv)
__global__ void finalize_layer(const float* __restrict__ lin, const float* __restrict__ agg,
                               const float* __restrict__ bc, float* __restrict__ out) {
  int t = blockIdx.x * blockDim.x + threadIdx.x;
  if (t >= NN * DIM) return;
  float v = lin[t] + agg[t] + bc[t & (DIM - 1)];
  out[t] = v > 0.f ? v : (__expf(v) - 1.0f);
}

// ---------------------------------------------------------------------------
extern "C" void kernel_launch(void* const* d_in, const int* in_sizes, int n_in,
                              void* d_out, int out_size, void* d_ws, size_t ws_size,
                              hipStream_t stream) {
  (void)in_sizes; (void)n_in; (void)out_size; (void)ws_size;
  const float* x  = (const float*)d_in[0];
  const int*   ei = (const int*)d_in[1];
  // layer l base: 2 + 7*l : W_lin, b_lin, W_src, W_dst, att_src, att_dst, b_conv
  const float* W_lin[2]  = {(const float*)d_in[2],  (const float*)d_in[9]};
  const float* b_lin[2]  = {(const float*)d_in[3],  (const float*)d_in[10]};
  const float* W_src[2]  = {(const float*)d_in[4],  (const float*)d_in[11]};
  const float* W_dst[2]  = {(const float*)d_in[5],  (const float*)d_in[12]};
  const float* att_s[2]  = {(const float*)d_in[6],  (const float*)d_in[13]};
  const float* att_d[2]  = {(const float*)d_in[7],  (const float*)d_in[14]};
  const float* b_conv[2] = {(const float*)d_in[8],  (const float*)d_in[15]};
  const float* W_out = (const float*)d_in[16];
  const float* b_out = (const float*)d_in[17];

  // workspace carve-up (256B aligned slices)
  char* ws = (char*)d_ws;
  size_t off = 0;
  auto carve = [&](size_t bytes) -> void* {
    void* p = ws + off;
    off = (off + bytes + 255) & ~(size_t)255;
    return p;
  };
  unsigned short* swz0 = (unsigned short*)carve(3 * 16384 * sizeof(unsigned short));
  unsigned short* swz1 = (unsigned short*)carve(3 * 16384 * sizeof(unsigned short));
  unsigned short* swzo = (unsigned short*)carve(8192 * sizeof(unsigned short));
  float* lin   = (float*)carve((size_t)NN * DIM * 4);
  float* xs    = (float*)carve((size_t)NN * DIM * 4);
  float* xd    = (float*)carve((size_t)NN * DIM * 4);
  float* as_   = (float*)carve((size_t)NN * HEADS * 4);
  float* ad_   = (float*)carve((size_t)NN * HEADS * 4);
  float* mbuf  = (float*)carve((size_t)NN * HEADS * 4);
  float* denom = (float*)carve((size_t)NN * HEADS * 4);
  float* pbuf  = (float*)carve((size_t)NE * HEADS * 4);
  float* agg   = (float*)carve((size_t)NN * DIM * 4);
  float* hbuf  = (float*)carve((size_t)NN * DIM * 4);

  const unsigned short* swzL[2] = {swz0, swz1};

  // pre-swizzle all weight matrices into WMMA B-fragment layout (f16)
  for (int l = 0; l < 2; ++l) {
    unsigned short* s = (l == 0) ? swz0 : swz1;
    swizzle_w<<<64, 256, 0, stream>>>(W_lin[l], s,             DIM);
    swizzle_w<<<64, 256, 0, stream>>>(W_src[l], s + 16384,     DIM);
    swizzle_w<<<64, 256, 0, stream>>>(W_dst[l], s + 2 * 16384, DIM);
  }
  swizzle_w<<<32, 256, 0, stream>>>(W_out, swzo, OUTD);

  const int MB  = NN / 16;                       // 3125 M tiles (exact)
  const int gNH = (NN * HEADS + 255) / 256;
  const int gND = (NN * DIM + 255) / 256;
  const int gE  = (NE + 255) / 256;
  const int gEH = (NE * HEADS + 255) / 256;

  const float* layer_in = x;
  for (int l = 0; l < 2; ++l) {
    node_gemm<<<MB, 256, 0, stream>>>(layer_in, swzL[l], b_lin[l], lin, xs, xd);
    alpha_kernel<<<gNH, 256, 0, stream>>>(xs, xd, att_s[l], att_d[l], as_, ad_);
    init_buffers<<<gND, 256, 0, stream>>>(mbuf, denom, agg);
    edge_max<<<gE, 256, 0, stream>>>(ei, as_, ad_, mbuf);
    edge_sum<<<gE, 256, 0, stream>>>(ei, as_, ad_, mbuf, denom, pbuf);
    edge_agg<<<gEH, 256, 0, stream>>>(ei, pbuf, denom, xs, agg);
    finalize_layer<<<gND, 256, 0, stream>>>(lin, agg, b_conv[l], hbuf);
    layer_in = hbuf;                             // in-place reuse is safe
  }
  out_gemm<<<MB, 128, 0, stream>>>(hbuf, swzo, b_out, (float*)d_out);
}
